// SCI_Net_2628519985760
// MI455X (gfx1250) — matrix operand order repaired and checked by reading, not compiled, not run on hardware
//
#include <hip/hip_runtime.h>

typedef __attribute__((ext_vector_type(16))) _Float16 v16h;
typedef __attribute__((ext_vector_type(8)))  _Float16 v8h;
typedef __attribute__((ext_vector_type(2)))  _Float16 v2h;
typedef __attribute__((ext_vector_type(8)))  float    v8f;

__device__ __forceinline__ v8f v8f_zero() {
    v8f z;
#pragma unroll
    for (int i = 0; i < 8; ++i) z[i] = 0.0f;
    return z;
}

__device__ __forceinline__ int iclamp(int v, int lo, int hi) {
    return v < lo ? lo : (v > hi ? hi : v);
}

// leaky relu as a single max: for f<0, 0.01f > f; for f>=0, f >= 0.01f
__device__ __forceinline__ float lrelu(float f) { return fmaxf(f, 0.01f * f); }

// CDNA5 has V_TANH_F32 (TRANS unit); fall back to OCML if builtin absent.
__device__ __forceinline__ float fast_tanh(float x) {
#if __has_builtin(__builtin_amdgcn_tanhf)
    return __builtin_amdgcn_tanhf(x);
#else
    return tanhf(x);
#endif
}

// argument is tanh output, range [-1,1]: no range fixup needed -> raw v_exp_f32
__device__ __forceinline__ float fast_exp(float x) {
#if __has_builtin(__builtin_amdgcn_exp2f)
    return __builtin_amdgcn_exp2f(x * 1.4426950408889634f);
#else
    return __expf(x);
#endif
}

// ---------------------------------------------------------------------------
// Pack conv weights into the CDNA5 16-bit A-matrix (16x32) lane/VGPR layout:
// element j (0..15) of lane L: v = j>>1, par = j&1,
//   k = (v>>2)*16 + (L>>4)*8 + (v&3)*2 + par ;  m = L & 15
// A1 table (conv1): slot = (lvl*4+widx)*2 + mtile, K semantic k = kt*8 + ci
//                   (24 real K, 25..31 zero-padded)
// A2 table (conv2): slot = (lvl*4+widx)*3 + kt,    K semantic k = ci (full 32)
//                   (rows co>=8 zero-padded)
// ---------------------------------------------------------------------------
__global__ void prep_weights_kernel(const float* __restrict__ w1,
                                    const float* __restrict__ w2,
                                    _Float16* __restrict__ A1t,
                                    _Float16* __restrict__ A2t) {
    const int N1 = 3 * 4 * 2 * 512;   // 24 slots * 32 lanes * 16 elems
    const int N2 = 3 * 4 * 3 * 512;   // 36 slots * 32 lanes * 16 elems
    int tid = blockIdx.x * blockDim.x + threadIdx.x;
    if (tid < N1) {
        int j = tid & 15;
        int lane = (tid >> 4) & 31;
        int slot = tid >> 9;
        int mtile = slot & 1;
        int lw = slot >> 1;                       // lvl*4 + widx
        int v = j >> 1, par = j & 1;
        int k = ((v >> 2) << 4) + ((lane >> 4) << 3) + ((v & 3) << 1) + par;
        int co = mtile * 16 + (lane & 15);
        float val = 0.0f;
        if (k < 24) {                             // k = kt*8 + ci
            int kt = k >> 3, ci = k & 7;
            val = w1[((lw * 32 + co) * 8 + ci) * 3 + kt];
        }
        A1t[tid] = (_Float16)val;
    } else if (tid < N1 + N2) {
        int t2 = tid - N1;
        int j = t2 & 15;
        int lane = (t2 >> 4) & 31;
        int slot = t2 >> 9;
        int kt = slot % 3;
        int lw = slot / 3;
        int v = j >> 1, par = j & 1;
        int ci = ((v >> 2) << 4) + ((lane >> 4) << 3) + ((v & 3) << 1) + par; // 0..31
        int co = lane & 15;
        float val = 0.0f;
        if (co < 8)
            val = w2[((lw * 8 + co) * 32 + ci) * 3 + kt];
        A2t[t2] = (_Float16)val;
    }
}

// ---------------------------------------------------------------------------
// One wave computes conv_op for a 64-output time tile of one sequence/branch,
// then applies the lifting combine.
//   mode 0 (scale phase):  p=0: F_even_s = F_even * exp(conv(F_odd , w[1]))
//                          p=1: F_odd_s  = F_odd  * exp(conv(F_even, w[3]))
//   mode 1 (update phase): p=0: new[2n]   = F_odd_s  - conv(F_odd_s , w[0])
//                          p=1: new[2n+1] = F_even_s + conv(F_even_s, w[2])
// conv_op = conv1(8->32,K=3, edge pad 2) -> leaky relu -> conv2(32->8,K=3) -> tanh
// ---------------------------------------------------------------------------
__global__ __launch_bounds__(256)
void lift_conv_kernel(const float* __restrict__ src,
                      float* __restrict__ sE,
                      float* __restrict__ sO,
                      float* __restrict__ outPrev,
                      const _Float16* __restrict__ A1t,
                      const _Float16* __restrict__ A2t,
                      int T2, int tilesPerSeq, int lvl, int mode) {
    // per-wave tiles; transposed layouts so B operands are contiguous vectors
    __shared__ __align__(16) _Float16 xt[8][86][8];   // [wave][pos][ci]; row 84 = zeros
    __shared__ __align__(16) _Float16 ht[8][80][40];  // [wave][pos][co]; stride 40 halves

    const int lane = threadIdx.x & 31;
    const int wid  = threadIdx.x >> 5;
    const int p    = blockIdx.y;

    int gw   = blockIdx.x * 8 + wid;
    int tile = gw % tilesPerSeq;
    int seq  = gw / tilesPerSeq;
    int t0   = tile * 64;

    const float* cin;
    const float* mul;
    float* outp;
    int rs, step, coff, moff, widx;
    float sign = 0.0f;
    if (mode == 0) {
        int T = T2 * 2;
        cin  = src + (size_t)seq * 8 * T;
        mul  = cin;
        rs   = T; step = 2; coff = p; moff = 1 - p;
        outp = (p ? sO : sE) + (size_t)seq * 8 * T2;
        widx = lvl * 4 + 1 + 2 * p;
    } else {
        cin  = (p ? sE : sO) + (size_t)seq * 8 * T2;
        mul  = cin;
        rs   = T2; step = 1; coff = 0; moff = 0;
        int outSeq = (seq >> 7) * 256 + p * 128 + (seq & 127); // (2n+q)*B + b
        outp = outPrev + (size_t)outSeq * 8 * T2;
        widx = lvl * 4 + 2 * p;
        sign = p ? 1.0f : -1.0f;
    }

    // get the epilogue's multiplier rows moving early (global_prefetch_b8)
#pragma unroll
    for (int r = 0; r < 8; ++r)
        __builtin_prefetch(mul + (size_t)r * rs + (size_t)t0 * step + moff, 0, 1);

    // --- A operands (pre-packed, 32B per lane per tile) ---
    v16h a1lo = *(const v16h*)(A1t + (size_t)((widx * 2 + 0) * 32 + lane) * 16);
    v16h a1hi = *(const v16h*)(A1t + (size_t)((widx * 2 + 1) * 32 + lane) * 16);
    v16h a2[3];
#pragma unroll
    for (int kt = 0; kt < 3; ++kt)
        a2[kt] = *(const v16h*)(A2t + (size_t)((widx * 3 + kt) * 32 + lane) * 16);

    // --- stage input halo [t0-2, t0+81] (edge-clamped) transposed into LDS ---
    for (int i = lane; i < 84 * 8; i += 32) {
        int pos = i >> 3, ci = i & 7;
        int t = iclamp(t0 - 2 + pos, 0, T2 - 1);
        xt[wid][pos][ci] = (_Float16)cin[(size_t)ci * rs + (size_t)t * step + coff];
    }
    if (lane < 8) xt[wid][84][lane] = (_Float16)0.0f;   // zero row for K padding

    const int n  = lane & 15;   // time column within 16-wide N tile
    const int kh = lane >> 4;   // K half (0: K 0-7,16-23; 1: K 8-15,24-31)
    const _Float16* xw = &xt[wid][0][0];
    _Float16* hw = &ht[wid][0][0];

    // --- GEMM1: hidden[32ch][80 pos]; K = kt*8+ci (24 real), leaky relu ---
    // B build: low half = xt row pos0+kh (kt 0 / kt 1); high half = row pos0+2
    // (kt 2) for kh=0, zero row for kh=1. Two ds_load_b128 per operand.
#pragma unroll
    for (int nt = 0; nt < 5; ++nt) {
        int pos0 = nt * 16 + n;
        v8h blo = *(const v8h*)(xw + (size_t)(pos0 + kh) * 8);
        int rhi = kh ? 84 : (pos0 + 2);
        v8h bhi = *(const v8h*)(xw + (size_t)rhi * 8);
        v16h b1 = __builtin_shufflevector(blo, bhi,
                                          0, 1, 2, 3, 4, 5, 6, 7,
                                          8, 9, 10, 11, 12, 13, 14, 15);
        v8f c0 = __builtin_amdgcn_wmma_f32_16x16x32_f16(false, a1lo, false, b1,
                                                        (short)0, v8f_zero(), false, false);
        v8f c1 = __builtin_amdgcn_wmma_f32_16x16x32_f16(false, a1hi, false, b1,
                                                        (short)0, v8f_zero(), false, false);
        // D layout: c0 -> co = r + 8*kh, c1 -> co = 16 + r + 8*kh.
        // Pack pairs; compiler merges the four v2h stores into ds_store_b128.
        _Float16* hrow = hw + (size_t)pos0 * 40;
#pragma unroll
        for (int rr = 0; rr < 4; ++rr) {
            v2h pk0;
            pk0[0] = (_Float16)lrelu(c0[2 * rr]);
            pk0[1] = (_Float16)lrelu(c0[2 * rr + 1]);
            *(v2h*)(hrow + kh * 8 + 2 * rr) = pk0;
            v2h pk1;
            pk1[0] = (_Float16)lrelu(c1[2 * rr]);
            pk1[1] = (_Float16)lrelu(c1[2 * rr + 1]);
            *(v2h*)(hrow + 16 + kh * 8 + 2 * rr) = pk1;
        }
    }

    // --- GEMM2: 8(out ch, pad 16) x 64(time); 3 taps, K = ci (full 32) ---
#pragma unroll
    for (int nt2 = 0; nt2 < 4; ++nt2) {
        v8f acc = v8f_zero();
#pragma unroll
        for (int kt = 0; kt < 3; ++kt) {
            v8h b2lo = *(const v8h*)(hw + (size_t)(nt2 * 16 + n + kt) * 40 + kh * 8);
            v8h b2hi = *(const v8h*)(hw + (size_t)(nt2 * 16 + n + kt) * 40 + 16 + kh * 8);
            v16h b2 = __builtin_shufflevector(b2lo, b2hi, 0, 1, 2, 3, 4, 5, 6, 7, 8, 9, 10, 11, 12, 13, 14, 15);
            acc = __builtin_amdgcn_wmma_f32_16x16x32_f16(false, a2[kt], false, b2,
                                                         (short)0, acc, false, false);
        }
        // D layout: lanes 0-15 -> co=r (real), lanes 16-31 -> co=r+8 (zero pad)
        if (kh == 0) {
            int t = t0 + nt2 * 16 + n;
#pragma unroll
            for (int r = 0; r < 8; ++r) {
                float y = fast_tanh(acc[r]);
                float m = mul[(size_t)r * rs + (size_t)t * step + moff];
                float o = (mode == 0) ? m * fast_exp(y) : m + sign * y;
                outp[(size_t)r * T2 + t] = o;
            }
        }
    }
}

// G[b,c,s] = prev3[bitrev3(s&7), b, c, s>>3] + x[b,c,s]   (closed-form _realign)
__global__ void gather_add_kernel(const float* __restrict__ prev3,
                                  const float* __restrict__ x,
                                  float* __restrict__ G, int total) {
    int tid = blockIdx.x * blockDim.x + threadIdx.x;
    if (tid >= total) return;
    int s  = tid & 8191;
    int bc = tid >> 13;                 // b*8 + c
    int b  = bc >> 3, c = bc & 7;
    int low = s & 7;
    int g = ((low & 1) << 2) | (low & 2) | ((low >> 2) & 1);
    int t = s >> 3;
    G[tid] = prev3[(((size_t)(g * 128 + b) * 8 + c) << 10) + t] + x[tid];
}

// H[row,h] = relu(sum_s G[row,s]*W[h,s] + b[h]),  row = b*C+c (1024 rows), K=8192
__global__ __launch_bounds__(256)
void fc1_kernel(const float* __restrict__ G, const float* __restrict__ W,
                const float* __restrict__ bias, float* __restrict__ H) {
    __shared__ float red[50][256];
    int row = blockIdx.x;
    int tid = threadIdx.x;
    const float* g = G + (size_t)row * 8192;
    float acc[50];
#pragma unroll
    for (int h = 0; h < 50; ++h) acc[h] = 0.0f;
    for (int s = tid; s < 8192; s += 256) {
        float gv = g[s];
#pragma unroll
        for (int h = 0; h < 50; ++h) acc[h] += gv * W[(size_t)h * 8192 + s];
    }
#pragma unroll
    for (int h = 0; h < 50; ++h) red[h][tid] = acc[h];
    __syncthreads();
    for (int off = 128; off > 0; off >>= 1) {
        if (tid < off) {
            for (int h = 0; h < 50; ++h) red[h][tid] += red[h][tid + off];
        }
        __syncthreads();
    }
    if (tid < 50) {
        float v = red[tid][0] + bias[tid];
        H[(size_t)row * 50 + tid] = v > 0.0f ? v : 0.0f;
    }
}

__global__ void fc2_kernel(const float* __restrict__ H, const float* __restrict__ W,
                           const float* __restrict__ bias, float* __restrict__ out) {
    int tid = blockIdx.x * blockDim.x + threadIdx.x;
    if (tid >= 1024 * 24) return;
    int row = tid / 24, o = tid - row * 24;
    const float* h = H + (size_t)row * 50;
    const float* w = W + (size_t)o * 50;
    float a = bias[o];
#pragma unroll
    for (int i = 0; i < 50; ++i) a += h[i] * w[i];
    out[tid] = a;
}

extern "C" void kernel_launch(void* const* d_in, const int* in_sizes, int n_in,
                              void* d_out, int out_size, void* d_ws, size_t ws_size,
                              hipStream_t stream) {
    const float* x   = (const float*)d_in[0];
    const float* w1  = (const float*)d_in[1];
    const float* w2  = (const float*)d_in[2];
    const float* f1w = (const float*)d_in[3];
    const float* f1b = (const float*)d_in[4];
    const float* f2w = (const float*)d_in[5];
    const float* f2b = (const float*)d_in[6];
    float* outp = (float*)d_out;

    // workspace layout (bytes):
    //   [0,   32MiB)  prevBuf  (B*C*S floats; in-place across levels)
    //   [32,  64MiB)  sE / sO scale buffers; reused as G after the levels
    //   [64MiB + 0)   A1 f16 table (24,576B) ; A2 f16 table (36,864B)
    //   [64MiB+64KiB) H buffer (1024*50 floats)
    char* ws = (char*)d_ws;
    float*    prevBuf = (float*)ws;
    float*    Gbuf    = (float*)(ws + (size_t)33554432);
    float*    sEb     = Gbuf;
    float*    sOb     = Gbuf + 4194304;
    _Float16* A1t     = (_Float16*)(ws + (size_t)67108864);
    _Float16* A2t     = A1t + 12288;
    float*    Hbuf    = (float*)(ws + (size_t)67108864 + 65536);

    prep_weights_kernel<<<(30720 + 255) / 256, 256, 0, stream>>>(w1, w2, A1t, A2t);

    const float* cur = x;
    for (int lvl = 0; lvl < 3; ++lvl) {
        int T2 = 4096 >> lvl;                 // per-branch length this level
        int tiles = T2 / 64;
        dim3 grid(1024, 2);                   // seqs*tiles == 8192 waves at every level
        lift_conv_kernel<<<grid, 256, 0, stream>>>((const float*)cur, sEb, sOb, prevBuf,
                                                   A1t, A2t, T2, tiles, lvl, 0);
        lift_conv_kernel<<<grid, 256, 0, stream>>>((const float*)cur, sEb, sOb, prevBuf,
                                                   A1t, A2t, T2, tiles, lvl, 1);
        cur = prevBuf;                        // in-place: phase 1 only reads sE/sO
    }

    gather_add_kernel<<<(8388608 + 255) / 256, 256, 0, stream>>>(prevBuf, x, Gbuf, 8388608);
    fc1_kernel<<<1024, 256, 0, stream>>>(Gbuf, f1w, f1b, Hbuf);
    fc2_kernel<<<(24576 + 255) / 256, 256, 0, stream>>>(Hbuf, f2w, f2b, outp);
}
